// MesoNet_40879498729141
// MI455X (gfx1250) — compile-verified, matrix operand order checked by reading, not run on hardware
//
#include <hip/hip_runtime.h>
#include <hip/hip_bf16.h>
#include <stdint.h>

// ---------------------------------------------------------------- constants
constexpr int NN   = 2048;   // nodes
constexpr int NE   = 8192;   // edges
constexpr int ED   = 10;     // edge feature dim
constexpr int EH   = 32;     // edge hidden
constexpr int HID  = 160;    // node hidden
constexpr int NT   = HID / 16;            // 10 N-tiles of 16
constexpr int KW   = (EH + 1) * HID / 32; // 165 K-steps (w2 + b2 row-block)
constexpr int KR   = HID / 32;            // 5 K-steps for root GEMM
constexpr int XST  = 168;                 // padded LDS row stride
constexpr int BSTEP = NT * 32 * 8;        // 2560 dwords of packed B per K-step
constexpr int WPB  = 2;                   // waves (edge blocks) per workgroup

typedef __attribute__((ext_vector_type(16))) __bf16 v16bf;
typedef __attribute__((ext_vector_type(8)))  float  v8f;

union BF16x16 { uint32_t u[8]; v16bf v; };

// ---------------------------------------------------------------- helpers
static __device__ __forceinline__ uint32_t bf16_rne(float f) {
    uint32_t u = __float_as_uint(f);
    return (u + 0x7FFFu + ((u >> 16) & 1u)) >> 16;
}
static __device__ __forceinline__ uint32_t pk_bf16(float a, float b) {
    return bf16_rne(a) | (bf16_rne(b) << 16);
}

// Build A-fragment (16-bit A 16x32 layout: lanes 0-15 -> K {0..7,16..23},
// lanes 16-31 -> K {8..15,24..31}; bf16 pair per VGPR), scaled by hv.
static __device__ __forceinline__ v16bf build_a(const float* __restrict__ xrow,
                                                float hv, int hi) {
    BF16x16 a;
    const float* p = xrow + (hi ? 8 : 0);
#pragma unroll
    for (int v = 0; v < 8; ++v) {
        int off = (v < 4) ? (2 * v) : (16 + 2 * (v - 4));
        float2 xy = *reinterpret_cast<const float2*>(p + off);
        a.u[v] = pk_bf16(hv * xy.x, hv * xy.y);
    }
    return a.v;
}

static __device__ __forceinline__ v8f wmma_bf16(v16bf a, v16bf b, v8f c) {
    return __builtin_amdgcn_wmma_f32_16x16x32_bf16(false, a, false, b,
                                                   (short)0, c, false, false);
}

// Async fill of one K-step B panel (BSTEP dwords) into LDS, spread over the
// whole workgroup: each thread copies (BSTEP*4/16)/blockDim 16-byte chunks.
static __device__ __forceinline__ void async_fill(const uint32_t* __restrict__ gbase,
                                                  uint32_t lds_byte_base, int tid) {
#pragma unroll
    for (int i = 0; i < BSTEP / (WPB * 32 * 4); ++i) {   // 10 chunks/thread
        const int c = i * (WPB * 32) + tid;
        const uint32_t lo = lds_byte_base + (uint32_t)c * 16u;
        const uint32_t vo = (uint32_t)c * 16u;
        asm volatile("global_load_async_to_lds_b128 %0, %1, %2"
                     :: "v"(lo), "v"(vo), "s"(gbase) : "memory");
    }
}
#define WAIT_ASYNC_PART() asm volatile("s_wait_asynccnt 0xa" ::: "memory")
#define WAIT_ASYNC_ALL()  asm volatile("s_wait_asynccnt 0x0" ::: "memory")

// ---------------------------------------------------------------- tiny kernels
__global__ void zero_f_kernel(float* __restrict__ p, int n) {
    int i = blockIdx.x * blockDim.x + threadIdx.x;
    if (i < n) p[i] = 0.f;
}

__global__ void count_kernel(const int* __restrict__ ei, float* __restrict__ cnt) {
    int e = blockIdx.x * blockDim.x + threadIdx.x;
    if (e < NE) {
        int d = ei[NE + e];
        __hip_atomic_fetch_add(cnt + d, 1.0f, __ATOMIC_RELAXED, __HIP_MEMORY_SCOPE_AGENT);
    }
}

// h[e,k] = relu(sum_d ea[e,d]*w1[d,k] + b1[k])
__global__ void edge_mlp_kernel(const float* __restrict__ ea,
                                const float* __restrict__ w1,
                                const float* __restrict__ b1,
                                float* __restrict__ h) {
    int idx = blockIdx.x * blockDim.x + threadIdx.x;   // NE*EH threads
    int e = idx >> 5, k = idx & 31;
    float s = b1[k];
    const float* er = ea + e * ED;
#pragma unroll
    for (int d = 0; d < ED; ++d) s += er[d] * w1[d * EH + k];
    h[idx] = fmaxf(s, 0.f);
}

// Pack a row-major [K x 160] f32 matrix (rows 0..rows1-1 from A1, rest from A2)
// into split bf16 B-fragment images:
//   dword index = s*2560 + t*256 + q*128 + lane*4 + d,  v = q*4 + d
// so each lane's fragment is two 16B chunks 512B apart (2-way-min DS access).
__global__ void pack_b_kernel(const float* __restrict__ A1, int rows1,
                              const float* __restrict__ A2, int ksteps,
                              uint32_t* __restrict__ out) {
    int idx = blockIdx.x * blockDim.x + threadIdx.x;
    if (idx >= ksteps * BSTEP) return;
    int d    = idx & 3;
    int lane = (idx >> 2) & 31;
    int q    = (idx >> 7) & 1;
    int rest = idx >> 8;
    int t    = rest % NT;
    int s    = rest / NT;
    int v    = q * 4 + d;
    int jb   = (v < 4) ? (2 * v) : (16 + 2 * (v - 4));
    int kk   = s * 32 + jb + (lane >> 4) * 8;        // K row pair (kk, kk+1)
    int n    = t * 16 + (lane & 15);
    const float* r0 = (kk     < rows1) ? (A1 + (size_t)kk * HID)
                                       : (A2 + (size_t)(kk - rows1) * HID);
    const float* r1 = (kk + 1 < rows1) ? (A1 + (size_t)(kk + 1) * HID)
                                       : (A2 + (size_t)(kk + 1 - rows1) * HID);
    out[idx] = pk_bf16(r0[n], r1[n]);
}

// ---------------------------------------------------------------- fused message GEMM
// 2 waves / workgroup, each wave owns a 16-edge block; the packed w2|b2 panel
// for each K-step is double-buffered in LDS via async-to-LDS copies shared by
// both waves. D = A' @ (w2|b2), A'[e,(k,i)] = h[e,k]*x[src_e,i] (k==EH -> bias).
__global__ __launch_bounds__(WPB * 32)
void msg_kernel(const float* __restrict__ x, const float* __restrict__ h,
                const uint32_t* __restrict__ Wp, const int* __restrict__ ei,
                float* __restrict__ summed) {
    __shared__ __align__(16) float    x_lds[WPB][16 * XST];
    __shared__ float                  h_lds[WPB][16 * (EH + 1)];
    __shared__ int                    sid[WPB][16], did[WPB][16];
    __shared__ __align__(16) uint32_t Bbuf[2][BSTEP];

    const int tid  = threadIdx.x;
    const int w    = tid >> 5;
    const int lane = tid & 31;
    const int e0   = (blockIdx.x * WPB + w) * 16;

    if (lane < 16) {
        sid[w][lane] = ei[e0 + lane];
        did[w][lane] = ei[NE + e0 + lane];
    }
    __syncthreads();

    for (int r = 0; r < 16; ++r) {
        const float* xr = x + (size_t)sid[w][r] * HID;
        for (int c = lane; c < HID; c += 32) x_lds[w][r * XST + c] = xr[c];
    }
    for (int r = 0; r < 16; ++r) {
        const float* hr = h + (size_t)(e0 + r) * EH;
        for (int c = lane; c < EH + 1; c += 32)
            h_lds[w][r * (EH + 1) + c] = (c < EH) ? hr[c] : 1.0f;
    }

    const uint32_t bb0 = (uint32_t)(uintptr_t)&Bbuf[0][0];
    const uint32_t bb1 = (uint32_t)(uintptr_t)&Bbuf[1][0];

    // prologue: fill panels for steps 0 and 1
    async_fill(Wp, bb0, tid);
    async_fill(Wp + BSTEP, bb1, tid);
    WAIT_ASYNC_PART();          // own part of fill(0) done
    __syncthreads();            // everyone's fill(0) done (+ x/h staged)

    v8f acc[NT];
#pragma unroll
    for (int t = 0; t < NT; ++t) acc[t] = v8f{0.f, 0.f, 0.f, 0.f, 0.f, 0.f, 0.f, 0.f};

    const int    Mrow  = lane & 15;
    const int    hi    = lane >> 4;
    const float* xbase = &x_lds[w][Mrow * XST];
    const float* hbase = &h_lds[w][Mrow * (EH + 1)];

    int s = 0;
    for (int k = 0; k <= EH; ++k) {
        const float hv = hbase[k];
#pragma unroll
        for (int ss = 0; ss < HID / 32; ++ss, ++s) {
            const int cur = s & 1;
            const v16bf a = build_a(xbase + ss * 32, hv, hi);
            const uint32_t* Bc = &Bbuf[cur][0];
#pragma unroll
            for (int t = 0; t < NT; ++t) {
                BF16x16 b;
                const uint4 q0 = *reinterpret_cast<const uint4*>(Bc + t * 256 + lane * 4);
                const uint4 q1 = *reinterpret_cast<const uint4*>(Bc + t * 256 + 128 + lane * 4);
                b.u[0] = q0.x; b.u[1] = q0.y; b.u[2] = q0.z; b.u[3] = q0.w;
                b.u[4] = q1.x; b.u[5] = q1.y; b.u[6] = q1.z; b.u[7] = q1.w;
                acc[t] = wmma_bf16(a, b.v, acc[t]);
            }
            __syncthreads();                        // all waves done reading Bbuf[cur]
            if (s + 2 < KW) {
                async_fill(Wp + (size_t)(s + 2) * BSTEP, cur ? bb1 : bb0, tid);
                WAIT_ASYNC_PART();                  // fill(s+1) done, fill(s+2) in flight
            } else {
                WAIT_ASYNC_ALL();                   // drain tail fills
            }
            __syncthreads();                        // fill(s+1) visible to all waves
        }
    }

    // scatter-add messages (C layout: lane -> N, vgpr r -> M = r (+8 for hi lanes))
    const int col  = lane & 15;
    const int mofs = hi * 8;
#pragma unroll
    for (int r = 0; r < 8; ++r) {
        const int dstn = did[w][r + mofs];
        float* outr = summed + (size_t)dstn * HID + col;
#pragma unroll
        for (int t = 0; t < NT; ++t)
            __hip_atomic_fetch_add(outr + t * 16, acc[t][r],
                                   __ATOMIC_RELAXED, __HIP_MEMORY_SCOPE_AGENT);
    }
}

// ---------------------------------------------------------------- node update
// out[n,:] = summed[n,:]/max(cnt[n],1) + x[n,:] @ root + bias ; optional ReLU
__global__ __launch_bounds__(32)
void node_kernel(const float* __restrict__ x, const uint32_t* __restrict__ Rp,
                 const float* __restrict__ summed, const float* __restrict__ cnt,
                 const float* __restrict__ bias, float* __restrict__ out,
                 int do_relu) {
    __shared__ __align__(16) float x_lds[16 * XST];

    const int lane = threadIdx.x;
    const int n0   = blockIdx.x * 16;

    for (int r = 0; r < 16; ++r) {
        const float* xr = x + (size_t)(n0 + r) * HID;
        for (int c = lane; c < HID; c += 32) x_lds[r * XST + c] = xr[c];
    }
    __syncthreads();

    v8f acc[NT];
#pragma unroll
    for (int t = 0; t < NT; ++t) acc[t] = v8f{0.f, 0.f, 0.f, 0.f, 0.f, 0.f, 0.f, 0.f};

    const int    Mrow  = lane & 15;
    const int    hi    = lane >> 4;
    const float* xbase = x_lds + Mrow * XST;

#pragma unroll
    for (int s = 0; s < KR; ++s) {
        const v16bf a = build_a(xbase + s * 32, 1.0f, hi);
        const uint32_t* bs = Rp + (size_t)s * BSTEP;
        if (s + 1 < KR)
            __builtin_prefetch(bs + BSTEP + lane * 8, 0, 1);
#pragma unroll
        for (int t = 0; t < NT; ++t) {
            BF16x16 b;
            const uint4 q0 = *reinterpret_cast<const uint4*>(bs + t * 256 + lane * 4);
            const uint4 q1 = *reinterpret_cast<const uint4*>(bs + t * 256 + 128 + lane * 4);
            b.u[0] = q0.x; b.u[1] = q0.y; b.u[2] = q0.z; b.u[3] = q0.w;
            b.u[4] = q1.x; b.u[5] = q1.y; b.u[6] = q1.z; b.u[7] = q1.w;
            acc[t] = wmma_bf16(a, b.v, acc[t]);
        }
    }

    const int col  = lane & 15;
    const int mofs = hi * 8;
#pragma unroll
    for (int r = 0; r < 8; ++r) {
        const int n   = n0 + r + mofs;
        const float inv = 1.0f / fmaxf(cnt[n], 1.0f);
        const float* sr = summed + (size_t)n * HID + col;
        float*       orow = out  + (size_t)n * HID + col;
#pragma unroll
        for (int t = 0; t < NT; ++t) {
            float v = acc[t][r] + sr[t * 16] * inv + bias[t * 16 + col];
            if (do_relu) v = fmaxf(v, 0.f);
            orow[t * 16] = v;
        }
    }
}

// ---------------------------------------------------------------- launcher
extern "C" void kernel_launch(void* const* d_in, const int* in_sizes, int n_in,
                              void* d_out, int out_size, void* d_ws, size_t ws_size,
                              hipStream_t stream) {
    const float* x_in   = (const float*)d_in[0];
    const float* ea     = (const float*)d_in[1];
    const float* w1_a   = (const float*)d_in[2];
    const float* b1_a   = (const float*)d_in[3];
    const float* w2_a   = (const float*)d_in[4];
    const float* b2_a   = (const float*)d_in[5];
    const float* root_a = (const float*)d_in[6];
    const float* bias_a = (const float*)d_in[7];
    const float* w1_b   = (const float*)d_in[8];
    const float* b1_b   = (const float*)d_in[9];
    const float* w2_b   = (const float*)d_in[10];
    const float* b2_b   = (const float*)d_in[11];
    const float* root_b = (const float*)d_in[12];
    const float* bias_b = (const float*)d_in[13];
    const int*   ei     = (const int*)d_in[14];
    float* out = (float*)d_out;

    auto au = [](size_t x) { return (x + 255) & ~size_t(255); };
    char* ws = (char*)d_ws;
    size_t off = 0;
    float*    h_a    = (float*)(ws + off);    off += au((size_t)NE * EH * 4);
    float*    h_b    = (float*)(ws + off);    off += au((size_t)NE * EH * 4);
    uint32_t* Wp_a   = (uint32_t*)(ws + off); off += au((size_t)KW * BSTEP * 4);
    uint32_t* Wp_b   = (uint32_t*)(ws + off); off += au((size_t)KW * BSTEP * 4);
    uint32_t* Rp_a   = (uint32_t*)(ws + off); off += au((size_t)KR * BSTEP * 4);
    uint32_t* Rp_b   = (uint32_t*)(ws + off); off += au((size_t)KR * BSTEP * 4);
    float*    cnt    = (float*)(ws + off);    off += au((size_t)NN * 4);
    float*    summed = (float*)(ws + off);    off += au((size_t)NN * HID * 4);
    float*    xb1    = (float*)(ws + off);    off += au((size_t)NN * HID * 4);
    float*    xb2    = (float*)(ws + off);    off += au((size_t)NN * HID * 4);
    (void)off; (void)ws_size; (void)n_in; (void)in_sizes; (void)out_size;

    zero_f_kernel<<<(NN + 255) / 256, 256, 0, stream>>>(cnt, NN);
    count_kernel<<<(NE + 255) / 256, 256, 0, stream>>>(ei, cnt);

    edge_mlp_kernel<<<NE * EH / 256, 256, 0, stream>>>(ea, w1_a, b1_a, h_a);
    edge_mlp_kernel<<<NE * EH / 256, 256, 0, stream>>>(ea, w1_b, b1_b, h_b);

    pack_b_kernel<<<(KW * BSTEP + 255) / 256, 256, 0, stream>>>(w2_a, EH * HID, b2_a, KW, Wp_a);
    pack_b_kernel<<<(KW * BSTEP + 255) / 256, 256, 0, stream>>>(w2_b, EH * HID, b2_b, KW, Wp_b);
    pack_b_kernel<<<(KR * BSTEP + 255) / 256, 256, 0, stream>>>(root_a, HID, nullptr, KR, Rp_a);
    pack_b_kernel<<<(KR * BSTEP + 255) / 256, 256, 0, stream>>>(root_b, HID, nullptr, KR, Rp_b);

    // layer 1
    zero_f_kernel<<<(NN * HID + 255) / 256, 256, 0, stream>>>(summed, NN * HID);
    msg_kernel<<<NE / (16 * WPB), WPB * 32, 0, stream>>>(x_in, h_a, Wp_a, ei, summed);
    node_kernel<<<NN / 16, 32, 0, stream>>>(x_in, Rp_a, summed, cnt, bias_a, xb1, 1);
    // layer 2
    zero_f_kernel<<<(NN * HID + 255) / 256, 256, 0, stream>>>(summed, NN * HID);
    msg_kernel<<<NE / (16 * WPB), WPB * 32, 0, stream>>>(xb1, h_b, Wp_b, ei, summed);
    node_kernel<<<NN / 16, 32, 0, stream>>>(xb1, Rp_b, summed, cnt, bias_b, xb2, 1);
    // layer 3
    zero_f_kernel<<<(NN * HID + 255) / 256, 256, 0, stream>>>(summed, NN * HID);
    msg_kernel<<<NE / (16 * WPB), WPB * 32, 0, stream>>>(xb2, h_b, Wp_b, ei, summed);
    node_kernel<<<NN / 16, 32, 0, stream>>>(xb2, Rp_b, summed, cnt, bias_b, out, 0);
}
